// SparKDensifiyBlock_79405355368959
// MI455X (gfx1250) — matrix operand order repaired and checked by reading, compiled, not run on hardware
//
#include <hip/hip_runtime.h>

typedef __attribute__((ext_vector_type(16))) __bf16       v16bf;
typedef __attribute__((ext_vector_type(8)))  float        v8f;
typedef __attribute__((ext_vector_type(4)))  unsigned int v4u;

#define C_CH   128
#define HW_    16384
#define D_OUT  128
#define NPT    32                      // positions per group (2 WMMA N-tiles)
#define LSTR   12                      // dwords per lane slot (8 used + 4 pad, 16B aligned)
#define BLKSTR (32 * LSTR)             // dwords per fragment block (one 32-lane operand)
#define FRAGDW (16 * BLKSTR)           // 16 blocks: 2 tiles x 4 chunks x {hi,lo}

static __device__ __forceinline__ unsigned int bfbits(float f) {
    return (unsigned int)__builtin_bit_cast(unsigned short, (__bf16)f);
}

struct frag32 { v4u a, b; };           // 32B, assembled from two 16B LDS loads

static __device__ __forceinline__ v16bf load_frag(const unsigned int* p) {
    frag32 fr;
    fr.a = *(const v4u*)p;
    fr.b = *(const v4u*)(p + 4);
    return __builtin_bit_cast(v16bf, fr);
}

__global__ __launch_bounds__(256) void spark_densify_wmma(
    const float* __restrict__ bcff,
    const unsigned char* __restrict__ cur_active,
    const float* __restrict__ gamma,
    const float* __restrict__ beta,
    const float* __restrict__ mtok,
    const float* __restrict__ Wmat,     // (D, C) flattened 1x1 conv weight
    const float* __restrict__ bias,
    float* __restrict__ out,
    int ngroups)
{
    // Double-buffered B fragments in exact WMMA operand layout.
    // block index blk = (tile*4 + q)*2 + hl; dword addr = blk*BLKSTR + lane*LSTR + vp
    __shared__ unsigned int fragU[2 * FRAGDW];              // 48 KB
    __shared__ float redS[2][8 * 32], redQ[2][8 * 32];
    __shared__ float rsS[2][32], rmS[2][32];
    __shared__ unsigned int actS[2][32];
    __shared__ float gvS[D_OUT], bvS[D_OUT], mcS[D_OUT];    // W·γ, W·β+bias, W·mtok+bias
    __shared__ float gammaS[C_CH], betaS[C_CH], mtokS[C_CH];

    const int tid  = threadIdx.x;           // 0..255
    const int lane = tid & 31;
    const int wave = tid >> 5;              // 0..7 -> which 16-row D tile
    const int mrow = lane & 15;
    const int gsel = lane >> 4;

    if (tid < C_CH) {
        gammaS[tid] = gamma[tid];
        betaS[tid]  = beta[tid];
        mtokS[tid]  = mtok[tid];
    }
    __syncthreads();

    // ---- once per block: per-d reduction vectors ----
    if (tid < D_OUT) {
        const float* wr = Wmat + tid * C_CH;
        float g = 0.f, bb = 0.f, m = 0.f;
#pragma unroll 8
        for (int c = 0; c < C_CH; ++c) {
            const float w = wr[c];
            g  += w * gammaS[c];
            bb += w * betaS[c];
            m  += w * mtokS[c];
        }
        gvS[tid] = g;
        bvS[tid] = bb + bias[tid];
        mcS[tid] = m + bias[tid];
    }

    // ---- once per block: this wave's 16x128 slice of (W ⊙ γ), split-bf16 A frags ----
    // A (16x32 16-bit), this code's k order (B packed to match): M = lane&15; VGPR v, half j -> K = 16*gsel+2*v+j
    const int drow = wave * 16 + mrow;
    v16bf Ahi[4], Alo[4];
#pragma unroll
    for (int q = 0; q < 4; ++q) {
#pragma unroll
        for (int v = 0; v < 8; ++v) {
#pragma unroll
            for (int j = 0; j < 2; ++j) {
                const int K = 16 * gsel + 2 * v + j;
                const int c = q * 32 + K;
                const float w  = Wmat[drow * C_CH + c] * gammaS[c];
                const __bf16 h = (__bf16)w;
                Ahi[q][2 * v + j] = h;
                Alo[q][2 * v + j] = (__bf16)(w - (float)h);
            }
        }
    }
    __syncthreads();

    // per-lane output-row constants (d = wave*16 + 8*gsel + r)
    const int d0 = wave * 16 + 8 * gsel;
    float gvr[8], bvr[8], mcr[8];
#pragma unroll
    for (int r = 0; r < 8; ++r) {
        gvr[r] = gvS[d0 + r];
        bvr[r] = bvS[d0 + r];
        mcr[r] = mcS[d0 + r];
    }

    int bb = 0;
    for (int grp = blockIdx.x; grp < ngroups; grp += gridDim.x) {
        const int n0  = grp * NPT;
        const int b   = n0 >> 14;           // / HW_
        const int hw0 = n0 & (HW_ - 1);
        const float* xb = bcff + (size_t)b * C_CH * HW_ + hw0;
        unsigned int* fb = fragU + bb * FRAGDW;

        // ---- phase 1: full-line coalesced loads (lane = position), stats partials,
        //      split-bf16 pack straight into fragment-layout LDS ----
        const int pp   = lane;              // position 0..31
        const int cg   = wave;              // dword slot this wave feeds (pairs 2cg,2cg+1)
        const int tile = pp >> 4;
        const int fl0  = pp & 15;
        float s = 0.f, ss = 0.f;
#pragma unroll
        for (int q = 0; q < 4; ++q) {
#pragma unroll
            for (int g2 = 0; g2 < 2; ++g2) {
                const int c0 = q * 32 + 16 * g2 + 2 * cg;
                const float x0 = xb[(size_t)c0 * HW_ + pp];        // 128B contiguous / wave
                const float x1 = xb[(size_t)(c0 + 1) * HW_ + pp];
                s  += x0 + x1;
                ss += x0 * x0 + x1 * x1;
                const __bf16 h0 = (__bf16)x0;
                const __bf16 h1 = (__bf16)x1;
                const float  r0 = x0 - (float)h0;
                const float  r1 = x1 - (float)h1;
                const unsigned int hiP =
                    ((unsigned int)__builtin_bit_cast(unsigned short, h1) << 16) |
                    (unsigned int)__builtin_bit_cast(unsigned short, h0);
                const unsigned int loP = (bfbits(r1) << 16) | bfbits(r0);
                const int fl  = g2 * 16 + fl0;
                const int blk = (tile * 4 + q) * 2;
                fb[(blk + 0) * BLKSTR + fl * LSTR + cg] = hiP;
                fb[(blk + 1) * BLKSTR + fl * LSTR + cg] = loP;
            }
        }
        redS[bb][cg * 32 + pp] = s;
        redQ[bb][cg * 32 + pp] = ss;
        if (tid < 32) actS[bb][tid] = cur_active[(size_t)b * HW_ + hw0 + tid];
        __syncthreads();

        if (tid < 32) {
            float sum = 0.f, sq = 0.f;
#pragma unroll
            for (int k = 0; k < 8; ++k) {
                sum += redS[bb][k * 32 + tid];
                sq  += redQ[bb][k * 32 + tid];
            }
            const float mu  = sum * (1.0f / C_CH);
            const float var = sq * (1.0f / C_CH) - mu * mu;   // biased, as F.layer_norm
            const float rs  = rsqrtf(var + 1e-6f);
            rsS[bb][tid] = rs;
            rmS[bb][tid] = rs * mu;
        }
        __syncthreads();

        // ---- phase 2: two N-tiles of WMMA from LDS fragments (A frags reused) ----
        v8f acc0 = {}, acc1 = {};
#pragma unroll
        for (int q = 0; q < 4; ++q) {
            const v16bf B0h = load_frag(&fb[((0 * 4 + q) * 2 + 0) * BLKSTR + lane * LSTR]);
            const v16bf B0l = load_frag(&fb[((0 * 4 + q) * 2 + 1) * BLKSTR + lane * LSTR]);
            acc0 = __builtin_amdgcn_wmma_f32_16x16x32_bf16(false, Alo[q], false, B0h,
                                                           (short)0, acc0, false, false);
            acc0 = __builtin_amdgcn_wmma_f32_16x16x32_bf16(false, Ahi[q], false, B0l,
                                                           (short)0, acc0, false, false);
            acc0 = __builtin_amdgcn_wmma_f32_16x16x32_bf16(false, Ahi[q], false, B0h,
                                                           (short)0, acc0, false, false);
            const v16bf B1h = load_frag(&fb[((1 * 4 + q) * 2 + 0) * BLKSTR + lane * LSTR]);
            const v16bf B1l = load_frag(&fb[((1 * 4 + q) * 2 + 1) * BLKSTR + lane * LSTR]);
            acc1 = __builtin_amdgcn_wmma_f32_16x16x32_bf16(false, Alo[q], false, B1h,
                                                           (short)0, acc1, false, false);
            acc1 = __builtin_amdgcn_wmma_f32_16x16x32_bf16(false, Ahi[q], false, B1l,
                                                           (short)0, acc1, false, false);
            acc1 = __builtin_amdgcn_wmma_f32_16x16x32_bf16(false, Ahi[q], false, B1h,
                                                           (short)0, acc1, false, false);
        }

        // ---- phase 3: fold LayerNorm scalars + mask fill, store both tiles ----
        {
            const float rs = rsS[bb][mrow];
            const float rm = rmS[bb][mrow];
            const bool act = actS[bb][mrow] != 0;
            float* ob = out + (size_t)b * D_OUT * HW_ + hw0 + mrow;
#pragma unroll
            for (int r = 0; r < 8; ++r) {
                const float val = rs * acc0[r] - rm * gvr[r] + bvr[r];
                ob[(size_t)(d0 + r) * HW_] = act ? val : mcr[r];
            }
        }
        {
            const int col = mrow + 16;
            const float rs = rsS[bb][col];
            const float rm = rmS[bb][col];
            const bool act = actS[bb][col] != 0;
            float* ob = out + (size_t)b * D_OUT * HW_ + hw0 + col;
#pragma unroll
            for (int r = 0; r < 8; ++r) {
                const float val = rs * acc1[r] - rm * gvr[r] + bvr[r];
                ob[(size_t)(d0 + r) * HW_] = act ? val : mcr[r];
            }
        }
        bb ^= 1;    // double buffer: no trailing barrier needed
    }
}

extern "C" void kernel_launch(void* const* d_in, const int* in_sizes, int n_in,
                              void* d_out, int out_size, void* d_ws, size_t ws_size,
                              hipStream_t stream) {
    const float*         bcff   = (const float*)d_in[0];
    const unsigned char* active = (const unsigned char*)d_in[1];
    const float*         gamma  = (const float*)d_in[2];
    const float*         beta   = (const float*)d_in[3];
    const float*         mtok   = (const float*)d_in[4];
    const float*         convw  = (const float*)d_in[5];
    const float*         convb  = (const float*)d_in[6];
    float*               out    = (float*)d_out;

    const int npos    = in_sizes[1];          // B*H*W = 524288
    const int ngroups = npos / NPT;           // 16384 groups of 32 positions

    dim3 grid(2048), block(256);
    spark_densify_wmma<<<grid, block, 0, stream>>>(
        bcff, active, gamma, beta, mtok, convw, convb, out, ngroups);
}
